// AttentionDecoderGRU_6433861009949
// MI455X (gfx1250) — matrix operand order repaired and checked by reading, not compiled, not run on hardware
//
#include <hip/hip_runtime.h>
#include <hip/hip_bf16.h>

// ---------------- types for WMMA ----------------
typedef __attribute__((ext_vector_type(16))) __bf16 v16bf;
typedef __attribute__((ext_vector_type(8)))  float  v8f;

union FragBF { uint4 q[2]; v16bf v; };

// ---------------- helpers ----------------
__device__ __forceinline__ float bf2f(unsigned short u) {
    unsigned int x = ((unsigned int)u) << 16;
    float f; __builtin_memcpy(&f, &x, 4); return f;
}
__device__ __forceinline__ unsigned short f2bf(float f) {
    unsigned int x; __builtin_memcpy(&x, &f, 4);
    unsigned int r = x + 0x7fffu + ((x >> 16) & 1u);   // round-to-nearest-even
    return (unsigned short)(r >> 16);
}
__device__ __forceinline__ float sigmoidf_(float x) { return 1.0f / (1.0f + __expf(-x)); }

__device__ __forceinline__ v8f wmma_bf16(const FragBF& a, const FragBF& b, v8f c) {
    return __builtin_amdgcn_wmma_f32_16x16x32_bf16(false, a.v, false, b.v,
                                                   (short)0, c, false, false);
}

// =================================================================
// 2x2-blocked WMMA GEMM:  D(M x N) = X(M x K) @ W(N x K)^T + bias
// X, W bf16 row-major; D f32 with row stride ldd.
// grid = (N/32, M/32), block = 32: one wave computes a 32x32 tile
// (4 accumulators), so 8 b128 loads feed 4 WMMAs per k-step.
// =================================================================
__global__ void adg_wmma_gemm_f32(const unsigned short* __restrict__ X,
                                  const unsigned short* __restrict__ W,
                                  const float* __restrict__ bias,
                                  float* __restrict__ D,
                                  int K, int ldd)
{
    const int lane = threadIdx.x;
    const int half = lane >> 4;
    const int l16  = lane & 15;
    const int n0 = blockIdx.x * 32 + l16;      // B/D column for nt0
    const int m0 = blockIdx.y * 32 + l16;      // A row for mt0

    const unsigned short* arow0 = X + (size_t)m0 * K;
    const unsigned short* arow1 = arow0 + (size_t)16 * K;
    const unsigned short* brow0 = W + (size_t)n0 * K + 16 * half;
    const unsigned short* brow1 = brow0 + (size_t)16 * K;

    v8f c00 = {}, c01 = {}, c10 = {}, c11 = {};
    for (int k0 = 0; k0 < K; k0 += 32) {
        FragBF a0, a1, b0, b1;
        a0.q[0] = *(const uint4*)(arow0 + k0 + 16 * half);
        a0.q[1] = *(const uint4*)(arow0 + k0 + 16 * half + 8);
        a1.q[0] = *(const uint4*)(arow1 + k0 + 16 * half);
        a1.q[1] = *(const uint4*)(arow1 + k0 + 16 * half + 8);
        b0.q[0] = *(const uint4*)(brow0 + k0);
        b0.q[1] = *(const uint4*)(brow0 + k0 + 8);
        b1.q[0] = *(const uint4*)(brow1 + k0);
        b1.q[1] = *(const uint4*)(brow1 + k0 + 8);
        c00 = wmma_bf16(a0, b0, c00);
        c01 = wmma_bf16(a0, b1, c01);
        c10 = wmma_bf16(a1, b0, c10);
        c11 = wmma_bf16(a1, b1, c11);
    }
    const float bs0 = bias[n0];
    const float bs1 = bias[n0 + 16];
#pragma unroll
    for (int j = 0; j < 8; ++j) {
        int r0 = blockIdx.y * 32 + 8 * half + j;
        D[(size_t)r0 * ldd + n0]            = c00[j] + bs0;
        D[(size_t)r0 * ldd + n0 + 16]       = c01[j] + bs1;
        D[(size_t)(r0 + 16) * ldd + n0]     = c10[j] + bs0;
        D[(size_t)(r0 + 16) * ldd + n0 + 16] = c11[j] + bs1;
    }
}

// Same but bf16 output (used for K/V caches).
__global__ void adg_wmma_gemm_bf16(const unsigned short* __restrict__ X,
                                   const unsigned short* __restrict__ W,
                                   const float* __restrict__ bias,
                                   unsigned short* __restrict__ D,
                                   int K, int ldd)
{
    const int lane = threadIdx.x;
    const int half = lane >> 4;
    const int l16  = lane & 15;
    const int n0 = blockIdx.x * 32 + l16;
    const int m0 = blockIdx.y * 32 + l16;

    const unsigned short* arow0 = X + (size_t)m0 * K;
    const unsigned short* arow1 = arow0 + (size_t)16 * K;
    const unsigned short* brow0 = W + (size_t)n0 * K + 16 * half;
    const unsigned short* brow1 = brow0 + (size_t)16 * K;

    v8f c00 = {}, c01 = {}, c10 = {}, c11 = {};
    for (int k0 = 0; k0 < K; k0 += 32) {
        FragBF a0, a1, b0, b1;
        a0.q[0] = *(const uint4*)(arow0 + k0 + 16 * half);
        a0.q[1] = *(const uint4*)(arow0 + k0 + 16 * half + 8);
        a1.q[0] = *(const uint4*)(arow1 + k0 + 16 * half);
        a1.q[1] = *(const uint4*)(arow1 + k0 + 16 * half + 8);
        b0.q[0] = *(const uint4*)(brow0 + k0);
        b0.q[1] = *(const uint4*)(brow0 + k0 + 8);
        b1.q[0] = *(const uint4*)(brow1 + k0);
        b1.q[1] = *(const uint4*)(brow1 + k0 + 8);
        c00 = wmma_bf16(a0, b0, c00);
        c01 = wmma_bf16(a0, b1, c01);
        c10 = wmma_bf16(a1, b0, c10);
        c11 = wmma_bf16(a1, b1, c11);
    }
    const float bs0 = bias[n0];
    const float bs1 = bias[n0 + 16];
#pragma unroll
    for (int j = 0; j < 8; ++j) {
        int r0 = blockIdx.y * 32 + 8 * half + j;
        D[(size_t)r0 * ldd + n0]             = f2bf(c00[j] + bs0);
        D[(size_t)r0 * ldd + n0 + 16]        = f2bf(c01[j] + bs1);
        D[(size_t)(r0 + 16) * ldd + n0]      = f2bf(c10[j] + bs0);
        D[(size_t)(r0 + 16) * ldd + n0 + 16] = f2bf(c11[j] + bs1);
    }
}

// =================================================================
// Pre/post-processing kernels
// =================================================================
__global__ void adg_cvt_bf16(const float* __restrict__ x, unsigned short* __restrict__ y, int n) {
    int i = blockIdx.x * 256 + threadIdx.x;
    if (i < n) y[i] = f2bf(x[i]);
}

__global__ void adg_zero_u16(unsigned short* __restrict__ p, int n) {
    int i = blockIdx.x * 256 + threadIdx.x;
    if (i < n) p[i] = 0;
}

// LayerNorm over C=512 of transposed feats; one block per (b,s) row.
__global__ void adg_ln_feat(const float* __restrict__ feats,     // (B, C=512, S=196)
                            const float* __restrict__ g,
                            const float* __restrict__ bta,
                            unsigned short* __restrict__ seqbf)  // (B*S, 512) bf16
{
    const int bs = blockIdx.x;                 // 0..6271
    const int b = bs / 196, s = bs - b * 196;
    const int t = threadIdx.x;                 // 256 threads, 2 channels each
    const float* base = feats + (size_t)b * 512 * 196 + s;
    float x0 = base[(size_t)t * 196];
    float x1 = base[(size_t)(t + 256) * 196];

    __shared__ float sh[256];
    sh[t] = x0 + x1; __syncthreads();
    for (int off = 128; off > 0; off >>= 1) { if (t < off) sh[t] += sh[t + off]; __syncthreads(); }
    float m = sh[0] * (1.0f / 512.0f);
    __syncthreads();
    float d0 = x0 - m, d1 = x1 - m;
    sh[t] = d0 * d0 + d1 * d1; __syncthreads();
    for (int off = 128; off > 0; off >>= 1) { if (t < off) sh[t] += sh[t + off]; __syncthreads(); }
    float rs = rsqrtf(sh[0] * (1.0f / 512.0f) + 1e-5f);

    unsigned short* o = seqbf + (size_t)bs * 512;
    o[t]       = f2bf(d0 * rs * g[t]       + bta[t]);
    o[t + 256] = f2bf(d1 * rs * g[t + 256] + bta[t + 256]);
}

// mean over S for each (b, c)
__global__ void adg_mean(const unsigned short* __restrict__ seqbf, float* __restrict__ mn) {
    int i = blockIdx.x * 256 + threadIdx.x;    // 16384
    int b = i >> 9, c = i & 511;
    const unsigned short* p = seqbf + (size_t)b * 196 * 512 + c;
    float s = 0.0f;
    for (int k = 0; k < 196; ++k) s += bf2f(p[(size_t)k * 512]);
    mn[i] = s * (1.0f / 196.0f);
}

// h0 = tanh(mean @ init_h_W^T + b)
__global__ void adg_h0(const float* __restrict__ mn, const float* __restrict__ Wf,
                       const float* __restrict__ bf_, float* __restrict__ h,
                       unsigned short* __restrict__ hbf) {
    int i = blockIdx.x * 256 + threadIdx.x;    // 16384
    int b = i >> 9, j = i & 511;
    const float* mr = mn + b * 512;
    const float* wr = Wf + (size_t)j * 512;
    float acc = bf_[j];
    for (int c = 0; c < 512; ++c) acc += mr[c] * wr[c];
    float v = tanhf(acc);
    h[i] = v; hbf[i] = f2bf(v);
}

// gather token embeddings for t = 0..30, convert to bf16
__global__ void adg_gather_emb(const float* __restrict__ emb, const int* __restrict__ ids,
                               unsigned short* __restrict__ tok) {
    int i = blockIdx.x * 256 + threadIdx.x;    // 32*31*256 = 253952
    if (i >= 32 * 31 * 256) return;
    int e = i & 255, bt = i >> 8, b = bt / 31, t = bt - b * 31;
    int id = ids[b * 32 + t];
    tok[i] = f2bf(emb[(size_t)id * 256 + e]);
}

// write x_t into gin[b][0:256]  (gin[b][256:768] holds ctx from previous step)
__global__ void adg_prep_x(const unsigned short* __restrict__ tok,
                           unsigned short* __restrict__ gin, int t) {
    int i = blockIdx.x * 256 + threadIdx.x;    // 8192
    int b = i >> 8, e = i & 255;
    gin[b * 768 + e] = tok[(size_t)(b * 31 + t) * 256 + e];
}

// GRU gates + state update
__global__ void adg_gru_elem(const float* __restrict__ gx, const float* __restrict__ gh,
                             float* __restrict__ h, unsigned short* __restrict__ hbf) {
    int i = blockIdx.x * 256 + threadIdx.x;    // 16384
    int b = i >> 9, j = i & 511;
    const float* gxr = gx + (size_t)b * 1536;
    const float* ghr = gh + (size_t)b * 1536;
    float r = sigmoidf_(gxr[j]        + ghr[j]);
    float z = sigmoidf_(gxr[512 + j]  + ghr[512 + j]);
    float n = tanhf   (gxr[1024 + j] + r * ghr[1024 + j]);
    float hn = (1.0f - z) * n + z * h[i];
    h[i] = hn; hbf[i] = f2bf(hn);
}

// attention: one block per (b, head); scores over S=196, softmax, context
__global__ void adg_attn(const float* __restrict__ q,                // (32, 512)
                         const unsigned short* __restrict__ Kb,      // (B*S, 512) bf16
                         const unsigned short* __restrict__ Vb,
                         unsigned short* __restrict__ ctx_bf)        // (32, 512) bf16
{
    const int b = blockIdx.x >> 3;
    const int hh = blockIdx.x & 7;
    const int t = threadIdx.x;                 // 256
    __shared__ float sc[196];
    __shared__ float inv_s;

    const float* qv = q + b * 512 + hh * 64;
    for (int s = t; s < 196; s += 256) {
        const unsigned short* kp = Kb + (size_t)(b * 196 + s) * 512 + hh * 64;
        float acc = 0.0f;
        for (int d = 0; d < 64; ++d) acc += qv[d] * bf2f(kp[d]);
        sc[s] = acc * 0.125f;                  // / sqrt(64)
    }
    __syncthreads();
    if (t == 0) {
        float mx = -1e30f;
        for (int s = 0; s < 196; ++s) mx = fmaxf(mx, sc[s]);
        float sum = 0.0f;
        for (int s = 0; s < 196; ++s) { float e = __expf(sc[s] - mx); sc[s] = e; sum += e; }
        inv_s = 1.0f / sum;
    }
    __syncthreads();
    if (t < 64) {
        float acc = 0.0f;
        for (int s = 0; s < 196; ++s)
            acc += sc[s] * bf2f(Vb[(size_t)(b * 196 + s) * 512 + hh * 64 + t]);
        ctx_bf[b * 512 + hh * 64 + t] = f2bf(acc * inv_s);
    }
}

// LN of attention output -> ctx, written as bf16 into gin[b][256:768]
__global__ void adg_ln_ctx(const float* __restrict__ obuf, const float* __restrict__ g,
                           const float* __restrict__ bta, unsigned short* __restrict__ gin) {
    const int b = blockIdx.x, t = threadIdx.x; // 32 blocks x 256
    const float* x = obuf + b * 512;
    float x0 = x[t], x1 = x[t + 256];
    __shared__ float sh[256];
    sh[t] = x0 + x1; __syncthreads();
    for (int off = 128; off > 0; off >>= 1) { if (t < off) sh[t] += sh[t + off]; __syncthreads(); }
    float m = sh[0] * (1.0f / 512.0f);
    __syncthreads();
    float d0 = x0 - m, d1 = x1 - m;
    sh[t] = d0 * d0 + d1 * d1; __syncthreads();
    for (int off = 128; off > 0; off >>= 1) { if (t < off) sh[t] += sh[t + off]; __syncthreads(); }
    float rs = rsqrtf(sh[0] * (1.0f / 512.0f) + 1e-5f);
    unsigned short* o = gin + b * 768 + 256;
    o[t]       = f2bf(d0 * rs * g[t]       + bta[t]);
    o[t + 256] = f2bf(d1 * rs * g[t + 256] + bta[t + 256]);
}

// =================================================================
// Host side
// =================================================================
extern "C" void kernel_launch(void* const* d_in, const int* in_sizes, int n_in,
                              void* d_out, int out_size, void* d_ws, size_t ws_size,
                              hipStream_t stream) {
    (void)in_sizes; (void)n_in; (void)out_size; (void)ws_size;

    const float* feats    = (const float*)d_in[0];
    const int*   tgt_ids  = (const int*)  d_in[1];
    const float* emb      = (const float*)d_in[2];
    const float* init_h_W = (const float*)d_in[3];
    const float* init_h_b = (const float*)d_in[4];
    const float* ln_f_g   = (const float*)d_in[5];
    const float* ln_f_b   = (const float*)d_in[6];
    const float* q_W      = (const float*)d_in[7];
    const float* q_b      = (const float*)d_in[8];
    const float* k_W      = (const float*)d_in[9];
    const float* k_b      = (const float*)d_in[10];
    const float* v_W      = (const float*)d_in[11];
    const float* v_b      = (const float*)d_in[12];
    const float* o_W      = (const float*)d_in[13];
    const float* o_b      = (const float*)d_in[14];
    const float* W_ih     = (const float*)d_in[15];
    const float* W_hh     = (const float*)d_in[16];
    const float* b_ih     = (const float*)d_in[17];
    const float* b_hh     = (const float*)d_in[18];
    const float* fc_W     = (const float*)d_in[19];
    const float* fc_b     = (const float*)d_in[20];
    const float* ln_a_g   = (const float*)d_in[21];
    const float* ln_a_b   = (const float*)d_in[22];
    float* out = (float*)d_out;                 // (B=32, T=31, V=32000)

    // ---- workspace carve-up ----
    char* ws = (char*)d_ws; size_t off = 0;
    auto alloc = [&](size_t bytes) -> void* {
        void* p = ws + off; off = (off + bytes + 255) & ~(size_t)255; return p;
    };
    unsigned short* seqbf = (unsigned short*)alloc((size_t)6272 * 512 * 2);
    unsigned short* Kbf   = (unsigned short*)alloc((size_t)6272 * 512 * 2);
    unsigned short* Vbf   = (unsigned short*)alloc((size_t)6272 * 512 * 2);
    unsigned short* kWbf  = (unsigned short*)alloc((size_t)512 * 512 * 2);
    unsigned short* vWbf  = (unsigned short*)alloc((size_t)512 * 512 * 2);
    unsigned short* qWbf  = (unsigned short*)alloc((size_t)512 * 512 * 2);
    unsigned short* oWbf  = (unsigned short*)alloc((size_t)512 * 512 * 2);
    unsigned short* Wihbf = (unsigned short*)alloc((size_t)1536 * 768 * 2);
    unsigned short* Whhbf = (unsigned short*)alloc((size_t)1536 * 512 * 2);
    unsigned short* fcWbf = (unsigned short*)alloc((size_t)32000 * 512 * 2);
    unsigned short* tokbf = (unsigned short*)alloc((size_t)32 * 31 * 256 * 2);
    float* meanb = (float*)alloc((size_t)32 * 512 * 4);
    float* hbuf  = (float*)alloc((size_t)32 * 512 * 4);
    unsigned short* hbf = (unsigned short*)alloc((size_t)32 * 512 * 2);
    unsigned short* gin = (unsigned short*)alloc((size_t)32 * 768 * 2);
    float* gx   = (float*)alloc((size_t)32 * 1536 * 4);
    float* gh   = (float*)alloc((size_t)32 * 1536 * 4);
    float* qbuf = (float*)alloc((size_t)32 * 512 * 4);
    unsigned short* cbuf = (unsigned short*)alloc((size_t)32 * 512 * 2);
    float* obuf = (float*)alloc((size_t)32 * 512 * 4);

    auto cvt = [&](const float* x, unsigned short* y, int n) {
        adg_cvt_bf16<<<(n + 255) / 256, 256, 0, stream>>>(x, y, n);
    };
    auto gemm_f32 = [&](const unsigned short* X, const unsigned short* W, const float* bias,
                        float* D, int M, int N, int K, int ldd) {
        adg_wmma_gemm_f32<<<dim3(N / 32, M / 32), 32, 0, stream>>>(X, W, bias, D, K, ldd);
    };
    auto gemm_bf16 = [&](const unsigned short* X, const unsigned short* W, const float* bias,
                         unsigned short* D, int M, int N, int K, int ldd) {
        adg_wmma_gemm_bf16<<<dim3(N / 32, M / 32), 32, 0, stream>>>(X, W, bias, D, K, ldd);
    };

    // ---- one-time preprocessing (per call; deterministic) ----
    adg_zero_u16<<<(32 * 768 + 255) / 256, 256, 0, stream>>>(gin, 32 * 768);  // ctx0 = 0
    cvt(k_W,  kWbf,  512 * 512);
    cvt(v_W,  vWbf,  512 * 512);
    cvt(q_W,  qWbf,  512 * 512);
    cvt(o_W,  oWbf,  512 * 512);
    cvt(W_ih, Wihbf, 1536 * 768);
    cvt(W_hh, Whhbf, 1536 * 512);
    cvt(fc_W, fcWbf, 32000 * 512);
    adg_gather_emb<<<(32 * 31 * 256 + 255) / 256, 256, 0, stream>>>(emb, tgt_ids, tokbf);
    adg_ln_feat<<<6272, 256, 0, stream>>>(feats, ln_f_g, ln_f_b, seqbf);
    adg_mean<<<64, 256, 0, stream>>>(seqbf, meanb);
    adg_h0<<<64, 256, 0, stream>>>(meanb, init_h_W, init_h_b, hbuf, hbf);
    gemm_bf16(seqbf, kWbf, k_b, Kbf, 6272, 512, 512, 512);
    gemm_bf16(seqbf, vWbf, v_b, Vbf, 6272, 512, 512, 512);

    // ---- 31 sequential decode steps ----
    for (int t = 0; t < 31; ++t) {
        adg_prep_x<<<32, 256, 0, stream>>>(tokbf, gin, t);
        gemm_f32(gin, Wihbf, b_ih, gx, 32, 1536, 768, 1536);
        gemm_f32(hbf, Whhbf, b_hh, gh, 32, 1536, 512, 1536);
        adg_gru_elem<<<64, 256, 0, stream>>>(gx, gh, hbuf, hbf);
        gemm_f32(hbf, qWbf, q_b, qbuf, 32, 512, 512, 512);
        adg_attn<<<256, 256, 0, stream>>>(qbuf, Kbf, Vbf, cbuf);
        gemm_f32(cbuf, oWbf, o_b, obuf, 32, 512, 512, 512);
        adg_ln_ctx<<<32, 256, 0, stream>>>(obuf, ln_a_g, ln_a_b, gin);
        // logits for step t -> out[b][t][:], row stride = 31*32000
        gemm_f32(hbf, fcWbf, fc_b, out + (size_t)t * 32000, 32, 32000, 512, 31 * 32000);
    }
}
